// GCN_618475290672
// MI455X (gfx1250) — matrix operand order repaired and checked by reading, not compiled, not run on hardware
//
#include <hip/hip_runtime.h>
#include <hip/hip_bf16.h>

typedef __attribute__((ext_vector_type(16))) _Float16 v16h;
typedef __attribute__((ext_vector_type(8)))  float    v8f;

#define F_H 40

// ---------------------------------------------------------------------------
// hardware fp32 global atomic add (no-return form, STOREcnt-tracked;
// s_endpgm's implicit wait-idle guarantees completion before kernel end)
// ---------------------------------------------------------------------------
__device__ __forceinline__ void gatomic_fadd(float* p, float v) {
    asm volatile("global_atomic_add_f32 %0, %1, off" :: "v"(p), "v"(v) : "memory");
}

// ---------------------------------------------------------------------------
// small utility kernels
// ---------------------------------------------------------------------------
__global__ void k_fill(float* __restrict__ p, float v, int n) {
    int i = blockIdx.x * blockDim.x + threadIdx.x;
    if (i < n) p[i] = v;
}

__global__ void k_deg_count(const int* __restrict__ dst, float* __restrict__ deg, int E) {
    int e = blockIdx.x * blockDim.x + threadIdx.x;
    if (e < E) gatomic_fadd(&deg[dst[e]], 1.0f);
}

__global__ void k_rsqrt_inplace(float* __restrict__ d, int n) {
    int i = blockIdx.x * blockDim.x + threadIdx.x;
    if (i < n) d[i] = rsqrtf(fmaxf(d[i], 1.0f));   // deg >= 1 always (self loop)
}

// ---------------------------------------------------------------------------
// Dense transform: hw[N,40] = act(hin) @ W  via v_wmma_f32_16x16x32_f16
// act = identity (layer 1) or relu(x + bprev) (layers 2/3, folds prev epilogue)
// One wave per 16-row tile; 3 column tiles cover 40 outputs; K chunks of 32.
// W is staged once per block into LDS, pre-swizzled into packed-f16 B-fragment
// layout, so the inner loop reads each fragment with 2x ds_load_b128, no guards.
// ---------------------------------------------------------------------------
template <int K, bool ACT>
__global__ void k_gemm_wmma(const float* __restrict__ hin,
                            const float* __restrict__ bprev,
                            const float* __restrict__ W,     // [K, 40] row-major
                            float* __restrict__ hw,          // [N, 40]
                            int N, int nTiles) {
    constexpr int NCHUNK = (K + 31) / 32;       // 1 (K=30) or 2 (K=40)
    constexpr int NWORDS = 3 * NCHUNK * 256;    // fragment words (u32 = f16 pair)

    // ---- stage B fragments in LDS (branch-free fill, runs once/block) ----
    __shared__ __align__(32) unsigned bfrag[NWORDS];
    for (int w = threadIdx.x; w < NWORDS; w += blockDim.x) {
        const int v   = w & 7;
        const int ln  = (w >> 3) & 31;
        const int cc  = w >> 8;                 // 0 .. 3*NCHUNK-1
        const int nt  = cc / NCHUNK;
        const int c   = cc - nt * NCHUNK;
        const int k0  = c * 32 + (v & 3) * 2 + (v >> 2) * 16 + ((ln >= 16) ? 8 : 0);
        const int col = nt * 16 + (ln & 15);
        const bool ok0 = (k0     < K) && (col < F_H);
        const bool ok1 = (k0 + 1 < K) && (col < F_H);
        float w0 = W[ok0 ? ((size_t)k0 * F_H + col)       : 0];
        float w1 = W[ok1 ? ((size_t)(k0 + 1) * F_H + col) : 0];
        w0 = ok0 ? w0 : 0.f;
        w1 = ok1 ? w1 : 0.f;
        union { _Float16 h[2]; unsigned u; } pk;
        pk.h[0] = (_Float16)w0;
        pk.h[1] = (_Float16)w1;
        bfrag[w] = pk.u;
    }
    __syncthreads();

    const int lane = threadIdx.x & 31;
    const int wave = threadIdx.x >> 5;
    const int tile = blockIdx.x * (blockDim.x >> 5) + wave;
    if (tile >= nTiles) return;                 // uniform per wave: EXEC stays all-ones

    const int  m0   = tile * 16;
    const bool hi   = lane >= 16;
    int rowL = m0 + (lane & 15);
    if (rowL >= N) rowL = N - 1;                // clamp for safe loads (store is guarded)
    const float* __restrict__ hrow = hin + (size_t)rowL * K;

    // ---- A fragments: 16x32 f16 per chunk (branch-free: clamp + select) --
    v16h a[NCHUNK];
#pragma unroll
    for (int c = 0; c < NCHUNK; ++c) {
        v16h av;
#pragma unroll
        for (int v = 0; v < 8; ++v) {
            const int k0  = c * 32 + (v & 3) * 2 + (v >> 2) * 16 + (hi ? 8 : 0);
            const int kk0 = (k0     < K) ? k0     : 0;
            const int kk1 = (k0 + 1 < K) ? k0 + 1 : 0;
            float x0 = hrow[kk0];
            float x1 = hrow[kk1];
            if (ACT) {
                x0 = fmaxf(x0 + bprev[kk0], 0.f);
                x1 = fmaxf(x1 + bprev[kk1], 0.f);
            }
            x0 = (k0     < K) ? x0 : 0.f;
            x1 = (k0 + 1 < K) ? x1 : 0.f;
            av[2 * v]     = (_Float16)x0;
            av[2 * v + 1] = (_Float16)x1;
        }
        a[c] = av;
    }

    // ---- 3 column tiles of 16 (covers 48, cols >= 40 zero-padded) -------
    const int  colBase  = lane & 15;
    const bool fullTile = (m0 + 15 < N);        // wave-uniform (true when N%16==0)
    const v16h* __restrict__ bfragv = (const v16h*)bfrag;   // 32B fragments
#pragma unroll
    for (int nt = 0; nt < 3; ++nt) {
        const int col = nt * 16 + colBase;
        v8f acc = {};
#pragma unroll
        for (int c = 0; c < NCHUNK; ++c) {
            const v16h bv = bfragv[(nt * NCHUNK + c) * 32 + lane];
            acc = __builtin_amdgcn_wmma_f32_16x16x32_f16(
                false, a[c], false, bv, (short)0, acc, false, false);
        }
        // ---- store D: lane holds col, VGPR v holds row m0 + v (+8 hi) ---
        if (fullTile) {
            // nt<2 -> col always < 40; nt==2 -> only colBase<8 valid (one branch)
            if (nt < 2 || colBase < 8) {
#pragma unroll
                for (int v = 0; v < 8; ++v) {
                    hw[(size_t)(m0 + v + (hi ? 8 : 0)) * F_H + col] = acc[v];
                }
            }
        } else {
#pragma unroll
            for (int v = 0; v < 8; ++v) {
                const int r = m0 + v + (hi ? 8 : 0);
                if (r < N && col < F_H) hw[(size_t)r * F_H + col] = acc[v];
            }
        }
    }
}

// ---------------------------------------------------------------------------
// acc[i] = dis[i]^2 * hw[i]   (self-loop term, also initializes acc)
// ---------------------------------------------------------------------------
__global__ void k_selfloop(const float4* __restrict__ hw4, const float* __restrict__ dis,
                           float4* __restrict__ acc4, int N) {
    unsigned tid = blockIdx.x * blockDim.x + threadIdx.x;
    if (tid >= (unsigned)N * 10u) return;
    const unsigned i = tid / 10u;
    const unsigned r = tid - i * 10u;
    const float s = dis[i] * dis[i];
    float4 m = hw4[(size_t)i * 10 + r];
    m.x *= s; m.y *= s; m.z *= s; m.w *= s;
    acc4[(size_t)i * 10 + r] = m;
}

// ---------------------------------------------------------------------------
// per-edge scatter: acc[dst] += dis[src]*dis[dst] * hw[src]
// 10 lanes per edge, float4 gathers, hw fp32 global atomics (L2 resident)
// ---------------------------------------------------------------------------
__global__ void k_edge_scatter(const float4* __restrict__ hw4, const float* __restrict__ dis,
                               const int* __restrict__ src, const int* __restrict__ dst,
                               float* __restrict__ acc, int E) {
    unsigned tid = blockIdx.x * blockDim.x + threadIdx.x;
    if (tid >= (unsigned)E * 10u) return;
    const unsigned e = tid / 10u;
    const unsigned r = tid - e * 10u;
    const int s = src[e];
    const int d = dst[e];
    const float nrm = dis[s] * dis[d];
    float4 m = hw4[(size_t)s * 10 + r];
    float* base = acc + (size_t)d * F_H + r * 4;
    gatomic_fadd(base + 0, m.x * nrm);
    gatomic_fadd(base + 1, m.y * nrm);
    gatomic_fadd(base + 2, m.z * nrm);
    gatomic_fadd(base + 3, m.w * nrm);
}

// ---------------------------------------------------------------------------
// pooled sums: psum[batch[i]] += relu(acc[i] + b3), cnt[batch[i]] += 1
// ---------------------------------------------------------------------------
__global__ void k_pool(const float4* __restrict__ acc4, const float* __restrict__ b3,
                       const int* __restrict__ batch,
                       float* __restrict__ psum, float* __restrict__ cnt, int N) {
    unsigned tid = blockIdx.x * blockDim.x + threadIdx.x;
    if (tid >= (unsigned)N * 10u) return;
    const unsigned i = tid / 10u;
    const unsigned r = tid - i * 10u;
    const int g = batch[i];
    float4 h = acc4[(size_t)i * 10 + r];
    const int f = (int)r * 4;
    float* p = psum + (size_t)g * F_H + f;
    gatomic_fadd(p + 0, fmaxf(h.x + b3[f + 0], 0.f));
    gatomic_fadd(p + 1, fmaxf(h.y + b3[f + 1], 0.f));
    gatomic_fadd(p + 2, fmaxf(h.z + b3[f + 2], 0.f));
    gatomic_fadd(p + 3, fmaxf(h.w + b3[f + 3], 0.f));
    if (r == 0) gatomic_fadd(&cnt[g], 1.0f);
}

// ---------------------------------------------------------------------------
// head: out[g] = (psum[g]/max(cnt,1)) @ Wl + bl     (G=1024, 40->2)
// ---------------------------------------------------------------------------
__global__ void k_final(const float* __restrict__ psum, const float* __restrict__ cnt,
                        const float* __restrict__ Wl, const float* __restrict__ bl,
                        float* __restrict__ out, int G) {
    int g = blockIdx.x * blockDim.x + threadIdx.x;
    if (g >= G) return;
    const float inv = 1.0f / fmaxf(cnt[g], 1.0f);
    float a0 = bl[0], a1 = bl[1];
#pragma unroll
    for (int k = 0; k < F_H; ++k) {
        const float h = psum[(size_t)g * F_H + k] * inv;
        a0 += h * Wl[k * 2 + 0];
        a1 += h * Wl[k * 2 + 1];
    }
    out[g * 2 + 0] = a0;
    out[g * 2 + 1] = a1;
}

// ---------------------------------------------------------------------------
extern "C" void kernel_launch(void* const* d_in, const int* in_sizes, int n_in,
                              void* d_out, int out_size, void* d_ws, size_t ws_size,
                              hipStream_t stream) {
    const float* x     = (const float*)d_in[0];
    const int*   ei    = (const int*)d_in[1];   // [2, E]: src row then dst row
    const int*   batch = (const int*)d_in[2];
    const float* W1    = (const float*)d_in[3];
    const float* b1    = (const float*)d_in[4];
    const float* W2    = (const float*)d_in[5];
    const float* b2    = (const float*)d_in[6];
    const float* W3    = (const float*)d_in[7];
    const float* b3    = (const float*)d_in[8];
    const float* Wl    = (const float*)d_in[9];
    const float* bl    = (const float*)d_in[10];
    float* out = (float*)d_out;

    const int N = in_sizes[0] / 30;
    const int E = in_sizes[1] / 2;
    const int G = out_size / 2;
    const int* src = ei;
    const int* dst = ei + E;

    // workspace carving (256B aligned regions)
    char* base = (char*)d_ws;
    size_t off = 0;
    auto carve = [&](size_t bytes) -> void* {
        void* p = base + off;
        off += (bytes + 255) & ~(size_t)255;
        return p;
    };
    float* dis  = (float*)carve((size_t)N * 4);            // deg -> dis in place
    float* hw   = (float*)carve((size_t)N * F_H * 4);      // dense-transform output
    float* acc  = (float*)carve((size_t)N * F_H * 4);      // aggregated output
    float* psum = (float*)carve((size_t)G * F_H * 4);
    float* cnt  = (float*)carve((size_t)G * 4);

    const int TB = 256;
    const int nTiles   = (N + 15) / 16;
    const int gemmBlk  = (nTiles + 7) / 8;                 // 8 waves / block
    const int nodeF4   = ((unsigned)N * 10u + TB - 1) / TB;
    const int edgeF4   = ((unsigned)E * 10u + TB - 1) / TB;

    // degree + symmetric normalization
    k_fill<<<(N + TB - 1) / TB, TB, 0, stream>>>(dis, 1.0f, N);          // self loops
    k_deg_count<<<(E + TB - 1) / TB, TB, 0, stream>>>(dst, dis, E);
    k_rsqrt_inplace<<<(N + TB - 1) / TB, TB, 0, stream>>>(dis, N);

    // layer 1: K=30, no input activation
    k_gemm_wmma<30, false><<<gemmBlk, TB, 0, stream>>>(x, nullptr, W1, hw, N, nTiles);
    k_selfloop<<<nodeF4, TB, 0, stream>>>((const float4*)hw, dis, (float4*)acc, N);
    k_edge_scatter<<<edgeF4, TB, 0, stream>>>((const float4*)hw, dis, src, dst, acc, E);

    // layer 2: reads relu(acc + b1) on the fly
    k_gemm_wmma<F_H, true><<<gemmBlk, TB, 0, stream>>>(acc, b1, W2, hw, N, nTiles);
    k_selfloop<<<nodeF4, TB, 0, stream>>>((const float4*)hw, dis, (float4*)acc, N);
    k_edge_scatter<<<edgeF4, TB, 0, stream>>>((const float4*)hw, dis, src, dst, acc, E);

    // layer 3
    k_gemm_wmma<F_H, true><<<gemmBlk, TB, 0, stream>>>(acc, b2, W3, hw, N, nTiles);
    k_selfloop<<<nodeF4, TB, 0, stream>>>((const float4*)hw, dis, (float4*)acc, N);
    k_edge_scatter<<<edgeF4, TB, 0, stream>>>((const float4*)hw, dis, src, dst, acc, E);

    // global mean pool + head
    k_fill<<<(G * F_H + TB - 1) / TB, TB, 0, stream>>>(psum, 0.0f, G * F_H);
    k_fill<<<(G + TB - 1) / TB, TB, 0, stream>>>(cnt, 0.0f, G);
    k_pool<<<nodeF4, TB, 0, stream>>>((const float4*)acc, b3, batch, psum, cnt, N);
    k_final<<<(G + TB - 1) / TB, TB, 0, stream>>>(psum, cnt, Wl, bl, out, G);
}
